// MixedLinear_QO_86139864088893
// MI455X (gfx1250) — 8-bit matrix operand order repaired and checked by reading, compiled, not run on hardware
//
#include <hip/hip_runtime.h>


typedef __attribute__((ext_vector_type(8))) int   v8i;
typedef __attribute__((__vector_size__(16)))  int b128_t;

#define MROWS   16384
#define KDIM    2048
#define NDIM    2048
#define KCH     64
#define NCHUNKS (KDIM / KCH)
#define BLK_M   128
#define BLK_N   128
#define LDSS    80

#if defined(__AMDGCN__) && __has_builtin(__builtin_amdgcn_global_load_async_to_lds_b128)
#define HAVE_ASYNC_LDS 1
#define NSTAGE 2
#else
#define HAVE_ASYNC_LDS 0
#define NSTAGE 1
#endif

typedef __attribute__((address_space(1))) b128_t* gas_b128_p;
typedef __attribute__((address_space(3))) b128_t* las_b128_p;

__device__ __forceinline__ void async_copy16(const signed char* g, signed char* l) {
#if HAVE_ASYNC_LDS
    signed char* gn = const_cast<signed char*>(g);
    __builtin_amdgcn_global_load_async_to_lds_b128(
        (gas_b128_p)gn, (las_b128_p)l,  0,  0);
#else
    (void)g; (void)l;
#endif
}

__device__ __forceinline__ void wait_async_all() {
#if HAVE_ASYNC_LDS
#if __has_builtin(__builtin_amdgcn_s_wait_asynccnt)
    __builtin_amdgcn_s_wait_asynccnt(0);
#else
    asm volatile("s_wait_asynccnt 0x0" ::: "memory");
#endif
#endif
}

__device__ __forceinline__ int quant1(float v, float rs, float lo, float hi) {
    return (int)rintf(fminf(fmaxf(v * rs, lo), hi));
}
__device__ __forceinline__ unsigned pack4(int a, int b, int c, int d) {
    return (a & 0xFF) | ((b & 0xFF) << 8) | ((c & 0xFF) << 16) | ((d & 0xFF) << 24);
}

__global__ __launch_bounds__(256)
void quant_kernel(const float* __restrict__ src, const float* __restrict__ scales,
                  unsigned* __restrict__ q4, unsigned* __restrict__ q8, int n4) {
    int t = blockIdx.x * 256 + threadIdx.x;
    if (t >= n4) return;
    const float4 v = ((const float4*)src)[t];
    const float rs0 = 1.0f / scales[0];
    const float rs1 = 1.0f / scales[1];
    q4[t] = pack4(quant1(v.x, rs0, -8.f, 7.f),  quant1(v.y, rs0, -8.f, 7.f),
                  quant1(v.z, rs0, -8.f, 7.f),  quant1(v.w, rs0, -8.f, 7.f));
    q8[t] = pack4(quant1(v.x, rs1, -128.f, 127.f), quant1(v.y, rs1, -128.f, 127.f),
                  quant1(v.z, rs1, -128.f, 127.f), quant1(v.w, rs1, -128.f, 127.f));
}

__global__ __launch_bounds__(512)
void mixgemm_kernel(const signed char* __restrict__ qx4, const signed char* __restrict__ qx8,
                    const signed char* __restrict__ qw4, const signed char* __restrict__ qw8,
                    const float* __restrict__ wmix, const float* __restrict__ a_scales,
                    const float* __restrict__ w_scales, const float* __restrict__ bias,
                    float* __restrict__ out) {
    __shared__ __align__(16) signed char sx[NSTAGE][2][BLK_M * LDSS];
    __shared__ __align__(16) signed char sw[NSTAGE][2][BLK_N * LDSS];

    const int tid  = threadIdx.x;
    const int wave = tid >> 5;
    const int lane = tid & 31;
    const int l15  = lane & 15;
    const int g    = lane >> 4;

    const int nb = blockIdx.x * BLK_N;
    const int mb = blockIdx.y * BLK_M;

    float Am[2] = {0.f, 0.f}, C1[2] = {0.f, 0.f}, C0[2] = {0.f, 0.f};
#pragma unroll
    for (int i = 0; i < 2; ++i)
#pragma unroll
        for (int j = 0; j < 2; ++j)
#pragma unroll
            for (int m = 0; m < 2; ++m)
#pragma unroll
                for (int n = 0; n < 2; ++n) {
                    const float wv = wmix[((i * 2 + j) * 2 + m) * 2 + n];
                    Am[m] += wv;
                    if (i) C1[n] += wv; else C0[n] += wv;
                }
    float Ffull[2][2], Fblk[2][2];
#pragma unroll
    for (int m = 0; m < 2; ++m)
#pragma unroll
        for (int n = 0; n < 2; ++n) {
            const float s = Am[m] * a_scales[m] * w_scales[n];
            Ffull[m][n] = s * C1[n];
            Fblk[m][n]  = s * C0[n];
        }
    const float W1    = C1[0] + C1[1];
    const float Wall  = W1 + C0[0] + C0[1];
    const bool  inBlk = (nb < 1024);
    const float bcoef = inBlk ? Wall : W1;

    const int wm = wave >> 2;
    const int wn = wave & 3;

    v8i zero8; for (int e = 0; e < 8; ++e) zero8[e] = 0;
    v8i accI[2][2][2][2];
#pragma unroll
    for (int a = 0; a < 2; ++a)
#pragma unroll
        for (int b = 0; b < 2; ++b)
#pragma unroll
            for (int c = 0; c < 2; ++c)
#pragma unroll
                for (int d = 0; d < 2; ++d) accI[a][b][c][d] = zero8;
    float accF[2][2][8];
#pragma unroll
    for (int a = 0; a < 2; ++a)
#pragma unroll
        for (int b = 0; b < 2; ++b)
#pragma unroll
            for (int e = 0; e < 8; ++e) accF[a][b][e] = 0.f;

    const int lr = tid >> 2;
    const int lc = (tid & 3) * 16;
    const signed char* gx[2] = {qx4, qx8};
    const signed char* gw[2] = {qw4, qw8};

    auto issue_stage = [&](int kc, int st) {
        const int kb = kc * KCH;
#pragma unroll
        for (int p = 0; p < 2; ++p) {
            async_copy16(gx[p] + (size_t)(mb + lr) * KDIM + kb + lc,
                         &sx[st][p][lr * LDSS + lc]);
            async_copy16(gw[p] + (size_t)(nb + lr) * KDIM + kb + lc,
                         &sw[st][p][lr * LDSS + lc]);
        }
    };

    auto compute_stage = [&](int st) {
        v8i afrag[2][2];
#pragma unroll
        for (int m = 0; m < 2; ++m)
#pragma unroll
            for (int mt = 0; mt < 2; ++mt) {
                const int row = wm * 32 + mt * 16 + l15;
                const signed char* bp = &sx[st][m][row * LDSS + g * 16];
                v8i a;
                ((uint2*)&a)[0] = *(const uint2*)(bp);
                ((uint2*)&a)[1] = *(const uint2*)(bp + 8);
                ((uint2*)&a)[2] = *(const uint2*)(bp + 32);
                ((uint2*)&a)[3] = *(const uint2*)(bp + 40);
                afrag[m][mt] = a;
            }
        v8i bfrag[2][2];
#pragma unroll
        for (int n = 0; n < 2; ++n)
#pragma unroll
            for (int nt = 0; nt < 2; ++nt) {
                const int row = wn * 32 + nt * 16 + l15;
                const signed char* bp = &sw[st][n][row * LDSS + g * 16];
                v8i b;
                ((int4*)&b)[0] = *(const int4*)(bp);
                ((int4*)&b)[1] = *(const int4*)(bp + 32);
                bfrag[n][nt] = b;
            }
#pragma unroll
        for (int mt = 0; mt < 2; ++mt)
#pragma unroll
            for (int nt = 0; nt < 2; ++nt)
#pragma unroll
                for (int m = 0; m < 2; ++m)
#pragma unroll
                    for (int n = 0; n < 2; ++n)
                        accI[mt][nt][m][n] = __builtin_amdgcn_wmma_i32_16x16x64_iu8(
                            true, afrag[m][mt], true, bfrag[n][nt],
                            accI[mt][nt][m][n], false, false);
    };

    auto fold = [&](const float F[2][2]) {
#pragma unroll
        for (int mt = 0; mt < 2; ++mt)
#pragma unroll
            for (int nt = 0; nt < 2; ++nt)
#pragma unroll
                for (int m = 0; m < 2; ++m)
#pragma unroll
                    for (int n = 0; n < 2; ++n)
#pragma unroll
                        for (int e = 0; e < 8; ++e)
                            accF[mt][nt][e] += F[m][n] * (float)accI[mt][nt][m][n][e];
    };

#if HAVE_ASYNC_LDS
    issue_stage(0, 0);
    for (int kc = 0; kc < NCHUNKS; ++kc) {
        const int cur = kc & 1;
        wait_async_all();
        __syncthreads();
        if (kc + 1 < NCHUNKS) issue_stage(kc + 1, cur ^ 1);
        compute_stage(cur);
        if (kc == 15 && inBlk) fold(Fblk);
    }
#else
    for (int kc = 0; kc < NCHUNKS; ++kc) {
        const int kb = kc * KCH;
#pragma unroll
        for (int p = 0; p < 2; ++p) {
            const uint4 vx = *(const uint4*)(gx[p] + (size_t)(mb + lr) * KDIM + kb + lc);
            *(uint4*)(&sx[0][p][lr * LDSS + lc]) = vx;
            const uint4 vw = *(const uint4*)(gw[p] + (size_t)(nb + lr) * KDIM + kb + lc);
            *(uint4*)(&sw[0][p][lr * LDSS + lc]) = vw;
        }
        __syncthreads();
        if (kc + 1 < NCHUNKS) {
#pragma unroll
            for (int p = 0; p < 2; ++p) {
                __builtin_prefetch(gx[p] + (size_t)(mb + lr) * KDIM + kb + KCH + lc, 0, 1);
                __builtin_prefetch(gw[p] + (size_t)(nb + lr) * KDIM + kb + KCH + lc, 0, 1);
            }
        }
        compute_stage(0);
        if (kc == 15 && inBlk) fold(Fblk);
        __syncthreads();
    }
#endif

    fold(Ffull);
#pragma unroll
    for (int mt = 0; mt < 2; ++mt)
#pragma unroll
        for (int nt = 0; nt < 2; ++nt) {
            const int col = nb + wn * 32 + nt * 16 + l15;
            const float bv = bias[col] * bcoef;
#pragma unroll
            for (int e = 0; e < 8; ++e) {
                const int row = mb + wm * 32 + mt * 16 + g * 8 + e;
                out[(size_t)row * NDIM + col] = accF[mt][nt][e] + bv;
            }
        }
}

extern "C" void kernel_launch(void* const* d_in, const int* in_sizes, int n_in,
                              void* d_out, int out_size, void* d_ws, size_t ws_size,
                              hipStream_t stream) {
    const float* x        = (const float*)d_in[0];
    const float* weights  = (const float*)d_in[1];
    const float* W        = (const float*)d_in[2];
    const float* b        = (const float*)d_in[3];
    const float* a_scales = (const float*)d_in[4];
    const float* w_scales = (const float*)d_in[5];
    float* out = (float*)d_out;

    const size_t Nx = (size_t)MROWS * KDIM;
    const size_t Nw = (size_t)NDIM  * KDIM;
    signed char* qx4 = (signed char*)d_ws;
    signed char* qx8 = qx4 + Nx;
    signed char* qw4 = qx8 + Nx;
    signed char* qw8 = qw4 + Nw;

    const int nx4 = (int)(Nx / 4), nw4 = (int)(Nw / 4);
    quant_kernel<<<nx4 / 256, 256, 0, stream>>>(x, a_scales, (unsigned*)qx4, (unsigned*)qx8, nx4);
    quant_kernel<<<nw4 / 256, 256, 0, stream>>>(W, w_scales, (unsigned*)qw4, (unsigned*)qw8, nw4);

    dim3 grid(NDIM / BLK_N, MROWS / BLK_M);
    mixgemm_kernel<<<grid, 512, 0, stream>>>(qx4, qx8, qw4, qw8,
                                             weights, a_scales, w_scales, b, out);
}
